// MXFP4LinearWithActivation_50139448213872
// MI455X (gfx1250) — matrix operand order repaired and checked by reading, not compiled, not run on hardware
//
#include <hip/hip_runtime.h>
#include <hip/hip_bf16.h>

typedef __attribute__((ext_vector_type(16))) __bf16 v16bf;
typedef __attribute__((ext_vector_type(8)))  __bf16 v8bf;
typedef __attribute__((ext_vector_type(8)))  float  v8f;
typedef __attribute__((ext_vector_type(4)))  int    v4i;

#define M_DIM 8192
#define K_DIM 2880
#define N_DIM 5760
#define NOUT  (N_DIM / 2)

#define BM 256
#define BN 128
#define BK 32
#define KTILES (K_DIM / BK)

#define AS1 __attribute__((address_space(1)))
#define AS3 __attribute__((address_space(3)))

// ---------------------------------------------------------------------------
// Async global -> LDS 16-byte copy (CDNA5 GLOBAL_LOAD_ASYNC_TO_LDS_B128,
// tracked by ASYNCcnt). Builtin confirmed present; params are v4i pointers
// in global / LDS address spaces.
// ---------------------------------------------------------------------------
__device__ __forceinline__ void async_copy16(const void* g, void* l) {
#if __has_builtin(__builtin_amdgcn_global_load_async_to_lds_b128)
    __builtin_amdgcn_global_load_async_to_lds_b128(
        (AS1 v4i*)g, (AS3 v4i*)l, 0, 0);
#else
    asm volatile("global_load_async_to_lds_b128 %0, %1, off"
                 :: "v"((unsigned)(uintptr_t)l),
                    "v"((unsigned long long)(uintptr_t)g)
                 : "memory");
#endif
}

__device__ __forceinline__ void wait_async0() {
#if __has_builtin(__builtin_amdgcn_s_wait_asynccnt)
    __builtin_amdgcn_s_wait_asynccnt(0);
#else
    asm volatile("s_wait_asynccnt 0x0" ::: "memory");
#endif
}

// ---------------------------------------------------------------------------
// Kernel 1: MXFP4 quantize+dequantize of W (K x N, bf16), writing the exact
// dequantized values TRANSPOSED as wqT (N x K, bf16). Dequantized MXFP4
// values are exactly representable in bf16 (2 mantissa bits needed).
// ---------------------------------------------------------------------------
__global__ void mxfp4_dequant_transpose(const __hip_bfloat16* __restrict__ w,
                                        __hip_bfloat16* __restrict__ wqT) {
    int idx = blockIdx.x * blockDim.x + threadIdx.x;
    const int total = (K_DIM / 32) * N_DIM;
    if (idx >= total) return;
    int n  = idx % N_DIM;   // consecutive threads -> consecutive n (coalesced)
    int kb = idx / N_DIM;

    const __hip_bfloat16* wp = w + (size_t)(kb * 32) * N_DIM + n;
    float vals[32];
    float amax = 0.0f;
#pragma unroll
    for (int i = 0; i < 32; ++i) {
        float v = __bfloat162float(wp[(size_t)i * N_DIM]);
        vals[i] = v;
        amax = fmaxf(amax, fabsf(v));
    }
    float am = (amax > 0.0f) ? amax : 1.0f;
    union { float f; unsigned u; } cv; cv.f = am;
    int e = (int)((cv.u >> 23) & 255u) - 127;      // floor(log2(am)), am normal
    float scale = ldexpf(1.0f, e - 2);             // E8M0 block scale
    float inv   = ldexpf(1.0f, -(e - 2));          // exact reciprocal

    __hip_bfloat16* op = wqT + (size_t)n * K_DIM + kb * 32;
    const float lut[8] = {0.0f, 0.5f, 1.0f, 1.5f, 2.0f, 3.0f, 4.0f, 6.0f};
#pragma unroll
    for (int i = 0; i < 32; ++i) {
        float y  = fminf(fmaxf(vals[i] * inv, -6.0f), 6.0f);
        float ay = fabsf(y);
        // searchsorted(mids, |y|, side='left') == count(mids strictly < |y|)
        int q = (ay > 0.25f) + (ay > 0.75f) + (ay > 1.25f) + (ay > 1.75f) +
                (ay > 2.5f)  + (ay > 3.5f)  + (ay > 5.0f);
        op[i] = __float2bfloat16(copysignf(lut[q], y) * scale);
    }
}

// ---------------------------------------------------------------------------
// Kernel 2: bf16 WMMA GEMM (x: M x K, wqT: N x K) + bias + fused swiglu.
// Block = 256 threads = 8 wave32s; block tile 256x128, K-step 32,
// double-buffered LDS filled by async-to-LDS copies.
// Wave grid 4x2: each wave owns 64x64 = 4x4 WMMA 16x16 f32 accumulators.
// ---------------------------------------------------------------------------
#define TILE_ELEMS (BM * BK + BN * BK)   // 12288 bf16 per buffer

__device__ __forceinline__ void issue_tile(__bf16* As, __bf16* Bs,
                                           const __hip_bfloat16* x,
                                           const __hip_bfloat16* wqT,
                                           int tileM, int tileN, int kb, int tid) {
    // A: 256x32 bf16 = 1024 16B chunks; B: 128x32 = 512 chunks; 256 threads.
#pragma unroll
    for (int i = 0; i < 4; ++i) {
        int c = tid + i * 256;
        int row = c >> 2, col = (c & 3) * 8;
        async_copy16(x + (size_t)(tileM + row) * K_DIM + kb + col,
                     As + row * BK + col);
    }
#pragma unroll
    for (int i = 0; i < 2; ++i) {
        int c = tid + i * 256;
        int row = c >> 2, col = (c & 3) * 8;
        async_copy16(wqT + (size_t)(tileN + row) * K_DIM + kb + col,
                     Bs + row * BK + col);
    }
}

__global__ __launch_bounds__(256)
void mxfp4_gemm_swiglu(const __hip_bfloat16* __restrict__ x,
                       const __hip_bfloat16* __restrict__ wqT,
                       const __hip_bfloat16* __restrict__ bias,
                       __hip_bfloat16* __restrict__ out) {
    __shared__ __align__(32) __bf16 smem[2 * TILE_ELEMS];

    const int tid  = threadIdx.x;
    const int lane = tid & 31;
    const int wave = tid >> 5;
    const int half = lane >> 4;   // 0 or 1
    const int l16  = lane & 15;

    const int tileN = blockIdx.x * BN;
    const int tileM = blockIdx.y * BM;
    const int waveM = (wave >> 1) * 64;   // 0/64/128/192
    const int waveN = (wave & 1)  * 64;   // 0/64

    v8f acc[4][4];
#pragma unroll
    for (int mi = 0; mi < 4; ++mi)
#pragma unroll
        for (int ni = 0; ni < 4; ++ni)
#pragma unroll
            for (int r = 0; r < 8; ++r) acc[mi][ni][r] = 0.0f;

    // Prologue: fill buffer 0.
    issue_tile(smem, smem + BM * BK, x, wqT, tileM, tileN, 0, tid);

    for (int kt = 0; kt < KTILES; ++kt) {
        int buf = kt & 1;
        __bf16* As = smem + buf * TILE_ELEMS;
        __bf16* Bs = As + BM * BK;

        wait_async0();        // my async copies (current tile) landed in LDS
        __syncthreads();      // everyone's landed; everyone done with buf^1

        if (kt + 1 < KTILES) {
            __bf16* An = smem + (buf ^ 1) * TILE_ELEMS;
            issue_tile(An, An + BM * BK, x, wqT, tileM, tileN,
                       (kt + 1) * BK, tid);
        }

        // A fragments (this code's contiguous k order, same as B):
        //   lane half h, element i -> K = half*16 + i, M = lane%16
        v16bf afrag[4];
#pragma unroll
        for (int mi = 0; mi < 4; ++mi) {
            int row = waveM + mi * 16 + l16;
            afrag[mi] = *(const v16bf*)&As[row * BK + half * 16];
        }
        // B fragments (contiguous k order, same as A): N = lane%16, K = half*16 + i
        v16bf bfrag[4];
#pragma unroll
        for (int ni = 0; ni < 4; ++ni) {
            int row = waveN + ni * 16 + l16;
            bfrag[ni] = *(const v16bf*)&Bs[row * BK + half * 16];
        }

#pragma unroll
        for (int mi = 0; mi < 4; ++mi)
#pragma unroll
            for (int ni = 0; ni < 4; ++ni)
                acc[mi][ni] = __builtin_amdgcn_wmma_f32_16x16x32_bf16(
                    false, afrag[mi], false, bfrag[ni],
                    (short)0, acc[mi][ni], false, false);
    }

    // Epilogue: h = acc + bias; swiglu pairs adjacent columns (even=a, odd=b).
    // C/D layout: VGPR r -> row M = r + half*8, col N = lane%16.
#pragma unroll
    for (int ni = 0; ni < 4; ++ni) {
        int colg = tileN + waveN + ni * 16 + l16;
        float bv = __bfloat162float(bias[colg]);
#pragma unroll
        for (int mi = 0; mi < 4; ++mi) {
#pragma unroll
            for (int r = 0; r < 8; ++r) {
                float h  = acc[mi][ni][r] + bv;
                float hp = __shfl_xor(h, 1, 32);   // partner column value
                if ((lane & 1) == 0) {
                    float a = fminf(h, 7.0f);
                    float b = fminf(fmaxf(hp, -7.0f), 7.0f);
                    float s = a / (1.0f + __expf(-1.702f * a));
                    float val = s * (b + 1.0f);
                    int rowg = tileM + waveM + mi * 16 + r + half * 8;
                    out[(size_t)rowg * NOUT + (colg >> 1)] = __float2bfloat16(val);
                }
            }
        }
    }
}

// ---------------------------------------------------------------------------
extern "C" void kernel_launch(void* const* d_in, const int* in_sizes, int n_in,
                              void* d_out, int out_size, void* d_ws, size_t ws_size,
                              hipStream_t stream) {
    const __hip_bfloat16* x    = (const __hip_bfloat16*)d_in[0];
    const __hip_bfloat16* wgt  = (const __hip_bfloat16*)d_in[1];
    const __hip_bfloat16* bias = (const __hip_bfloat16*)d_in[2];
    __hip_bfloat16* out = (__hip_bfloat16*)d_out;
    __hip_bfloat16* wqT = (__hip_bfloat16*)d_ws;   // N_DIM * K_DIM bf16 = ~33 MB

    int total = (K_DIM / 32) * N_DIM;
    mxfp4_dequant_transpose<<<(total + 255) / 256, 256, 0, stream>>>(wgt, wqT);

    dim3 grid(N_DIM / BN, M_DIM / BM);   // 45 x 32
    mxfp4_gemm_swiglu<<<grid, 256, 0, stream>>>(x, wqT, bias, out);
}
